// SpaceTimeTransformer_5669356834913
// MI455X (gfx1250) — matrix operand order repaired and checked by reading, compiled, not run on hardware
//
#include <hip/hip_runtime.h>
#include <hip/hip_bf16.h>

typedef __attribute__((ext_vector_type(16))) __bf16 v16bf;
typedef __attribute__((ext_vector_type(8)))  float  v8f;
typedef __attribute__((ext_vector_type(8)))  __bf16 bf16x8;

#define D        768
#define H_HEADS  12
#define DH       64
#define NP       196
#define NTOK     785
#define BATCH    2
#define FR       4
#define NLAYER   12
#define MLP_HID  3072
#define NCLS     1000
#define MTOK     (BATCH * NTOK)   /* 1570 */
#define QKVW     (3 * D)          /* 2304 */

#define BM 64
#define BN 64
#define BK 64

__device__ __forceinline__ __bf16 f2bf(float f) {
  unsigned u = __builtin_bit_cast(unsigned, f);
  unsigned r = u + 0x7FFFu + ((u >> 16) & 1u);   // round-to-nearest-even
  unsigned short hs = (unsigned short)(r >> 16);
  return __builtin_bit_cast(__bf16, hs);
}

// ---- CDNA5 async global->LDS copy (ASYNCcnt-tracked DMA), ISA §15.18.3 op 98.
// Generic shared-aperture pointers keep the LDS byte offset in the low 32 bits
// (ISA §10.2 LDS aperture), so truncation yields the VDST LDS-address operand.
__device__ __forceinline__ void async_copy16(const void* g, void* lds) {
  unsigned l = (unsigned)(unsigned long long)lds;
  asm volatile("global_load_async_to_lds_b128 %0, %1, off"
               :: "v"(l), "v"(g)
               : "memory");
}
__device__ __forceinline__ void async_wait0() {
  asm volatile("s_wait_asynccnt 0" ::: "memory");
}

// ---------------------------------------------------------------- cvt f32->bf16
__global__ __launch_bounds__(256) void cvt_k(const float* __restrict__ in,
                                             __bf16* __restrict__ out,
                                             long long n) {
  long long i = (long long)blockIdx.x * 256 + threadIdx.x;
  if (i < n) out[i] = f2bf(in[i]);
}

// ---------------------------------------------------------------- im2col (patch embed A matrix, bf16)
// A[m, c*256 + i*16 + j] = x[bf, c, hp*16+i, wp*16+j], m = bf*196 + hp*14 + wp
__global__ __launch_bounds__(256) void im2col_k(const float* __restrict__ x,
                                                __bf16* __restrict__ out) {
  long long i = (long long)blockIdx.x * 256 + threadIdx.x;
  const long long total = (long long)BATCH * FR * NP * D;
  if (i >= total) return;
  int cc = (int)(i % D);
  long long m = i / D;
  int c  = cc >> 8;
  int rm = cc & 255;
  int ii = rm >> 4;
  int jj = rm & 15;
  int bf = (int)(m / NP);
  int pp = (int)(m % NP);
  int hp = pp / 14, wp = pp % 14;
  long long xi = (((long long)bf * 3 + c) * 224 + (hp * 16 + ii)) * 224 + (wp * 16 + jj);
  out[i] = f2bf(x[xi]);
}

// ---------------------------------------------------------------- assemble h = [cls|tokens] + pos + temporal
__global__ __launch_bounds__(256) void assemble_h_k(const float* __restrict__ pout,
                                                    const float* __restrict__ cls,
                                                    const float* __restrict__ pos,
                                                    const float* __restrict__ temb,
                                                    float* __restrict__ h) {
  long long i = (long long)blockIdx.x * 256 + threadIdx.x;
  const long long total = (long long)MTOK * D;
  if (i >= total) return;
  int c = (int)(i % D);
  long long t = i / D;
  int tok = (int)(t % NTOK);
  int b   = (int)(t / NTOK);
  float val;
  if (tok == 0) {
    val = cls[c] + pos[c];
  } else {
    int tt = tok - 1;
    int f = tt / NP;
    int p = tt % NP;
    val = pout[(((long long)(b * FR + f)) * NP + p) * D + c]
        + pos[(long long)(1 + p) * D + c]
        + temb[(long long)f * D + c];
  }
  h[i] = val;
}

// ---------------------------------------------------------------- layernorm (fp32 in, bf16 out)
__global__ __launch_bounds__(256) void layernorm_k(const float* __restrict__ in,
                                                   const float* __restrict__ w,
                                                   const float* __restrict__ bi,
                                                   __bf16* __restrict__ out) {
  int row = blockIdx.x;
  int tid = threadIdx.x;
  const float* x = in + (long long)row * D;
  __shared__ float red[256];
  float s = 0.f;
  for (int c = tid; c < D; c += 256) s += x[c];
  red[tid] = s; __syncthreads();
  for (int st = 128; st > 0; st >>= 1) { if (tid < st) red[tid] += red[tid + st]; __syncthreads(); }
  float mu = red[0] * (1.f / D); __syncthreads();
  float v = 0.f;
  for (int c = tid; c < D; c += 256) { float dv = x[c] - mu; v += dv * dv; }
  red[tid] = v; __syncthreads();
  for (int st = 128; st > 0; st >>= 1) { if (tid < st) red[tid] += red[tid + st]; __syncthreads(); }
  float rs = rsqrtf(red[0] * (1.f / D) + 1e-6f);
  for (int c = tid; c < D; c += 256)
    out[(long long)row * D + c] = f2bf((x[c] - mu) * rs * w[c] + bi[c]);
}

// ---------------------------------------------------------------- WMMA GEMM: C[M,N] = A[M,K] * W[N,K]^T
// 64x64x64 block tile, async-DMA double-buffered LDS staging, bf16 WMMA f32 acc.
// epilogue: +bias, optional exact GELU, optional fp32 residual, fp32 and/or bf16 out.
// Requires K % 64 == 0 (all K here are 768 or 3072).
__global__ __launch_bounds__(256) void gemm_bf16_k(const __bf16* __restrict__ A, long long lda,
                                                   const __bf16* __restrict__ W,
                                                   const float* __restrict__ bias,
                                                   const float* __restrict__ residual,
                                                   float* __restrict__ outF,
                                                   __bf16* __restrict__ outB,
                                                   int M, int N, int K, int gelu) {
  __shared__ alignas(16) __bf16 As[2][BM * BK];
  __shared__ alignas(16) __bf16 Bs[2][BN * BK];

  const int tid  = threadIdx.x;
  const int lane = tid & 31;
  const int wave = tid >> 5;
  const int r    = lane & 15;
  const int half = lane >> 4;
  const int m0 = (wave & 3) * 16;     // wave tile inside block tile
  const int n0 = (wave >> 2) * 32;
  const int blockM = blockIdx.y * BM;
  const int blockN = blockIdx.x * BN;

  // Stage one 64x64 bf16 tile pair: 512 16B chunks each, 2 chunks/thread/operand.
  auto stage = [&](int buf, int k0) {
#pragma unroll
    for (int c = 0; c < 2; c++) {
      int idx = tid + c * 256;          // [0,512)
      int row = idx >> 3;               // 8 chunks per 64-elem row
      int kc  = (idx & 7) * 8;          // element offset within row
      int ar = blockM + row; if (ar >= M) ar = M - 1;
      async_copy16(A + (long long)ar * lda + k0 + kc, &As[buf][row * BK + kc]);
      int br = blockN + row; if (br >= N) br = N - 1;
      async_copy16(W + (long long)br * K + k0 + kc, &Bs[buf][row * BK + kc]);
    }
  };

  v8f acc0 = {0.f, 0.f, 0.f, 0.f, 0.f, 0.f, 0.f, 0.f};
  v8f acc1 = {0.f, 0.f, 0.f, 0.f, 0.f, 0.f, 0.f, 0.f};

  stage(0, 0);
  async_wait0();
  __syncthreads();

  int buf = 0;
  for (int k0 = 0; k0 < K; k0 += BK) {
    if (k0 + BK < K) stage(buf ^ 1, k0 + BK);   // DMA next slab while computing

#pragma unroll
    for (int ks = 0; ks < BK; ks += 32) {
      // A frag: lane half 0 -> K [0..15], half 1 -> K [16..31]
      const __bf16* ap  = &As[buf][(m0 + r) * BK + ks + half * 16];
      bf16x8 alo = *(const bf16x8*)ap;
      bf16x8 ahi = *(const bf16x8*)(ap + 8);
      // B frag: lane half 0 -> K [0..15], half 1 -> K [16..31]
      const __bf16* bp0 = &Bs[buf][(n0 + r)      * BK + ks + half * 16];
      const __bf16* bp1 = &Bs[buf][(n0 + 16 + r) * BK + ks + half * 16];
      bf16x8 b0lo = *(const bf16x8*)bp0;
      bf16x8 b0hi = *(const bf16x8*)(bp0 + 8);
      bf16x8 b1lo = *(const bf16x8*)bp1;
      bf16x8 b1hi = *(const bf16x8*)(bp1 + 8);
      v16bf a, b0, b1;
#pragma unroll
      for (int i = 0; i < 8; i++) {
        a[i]  = alo[i];  a[i + 8]  = ahi[i];
        b0[i] = b0lo[i]; b0[i + 8] = b0hi[i];
        b1[i] = b1lo[i]; b1[i + 8] = b1hi[i];
      }
      acc0 = __builtin_amdgcn_wmma_f32_16x16x32_bf16(false, a, false, b0, (short)0, acc0, false, false);
      acc1 = __builtin_amdgcn_wmma_f32_16x16x32_bf16(false, a, false, b1, (short)0, acc1, false, false);
    }

    async_wait0();     // this wave's copies into buf^1 landed in LDS
    __syncthreads();   // everyone done reading buf, everyone's copies visible
    buf ^= 1;
  }

#pragma unroll
  for (int i = 0; i < 8; i++) {
    int m = blockM + m0 + i + 8 * half;
    if (m >= M) continue;
    {
      int n = blockN + n0 + r;
      if (n < N) {
        float v = acc0[i];
        if (bias)     v += bias[n];
        if (gelu)     v = 0.5f * v * (1.0f + erff(v * 0.70710678118654752f));
        if (residual) v += residual[(long long)m * N + n];
        if (outF)     outF[(long long)m * N + n] = v;
        if (outB)     outB[(long long)m * N + n] = f2bf(v);
      }
    }
    {
      int n = blockN + n0 + 16 + r;
      if (n < N) {
        float v = acc1[i];
        if (bias)     v += bias[n];
        if (gelu)     v = 0.5f * v * (1.0f + erff(v * 0.70710678118654752f));
        if (residual) v += residual[(long long)m * N + n];
        if (outF)     outF[(long long)m * N + n] = v;
        if (outB)     outB[(long long)m * N + n] = f2bf(v);
      }
    }
  }
}

// ---------------------------------------------------------------- cls attention: query token0, keys = all 785
__global__ __launch_bounds__(64) void attn_cls_k(const float* __restrict__ qkv,
                                                 __bf16* __restrict__ out) {
  int h = blockIdx.x % H_HEADS;
  int b = blockIdx.x / H_HEADS;
  int tid = threadIdx.x;
  __shared__ float qs[DH];
  __shared__ float sc[NTOK];
  __shared__ float red[64];
  const float* base = qkv + (long long)b * NTOK * QKVW;
  qs[tid] = base[h * DH + tid] * 0.125f;
  __syncthreads();
  float lmax = -3.4e38f;
  for (int j = tid; j < NTOK; j += 64) {
    const float* kp = base + (long long)j * QKVW + D + h * DH;
    float s = 0.f;
#pragma unroll 8
    for (int dd = 0; dd < DH; dd++) s += qs[dd] * kp[dd];
    sc[j] = s;
    lmax = fmaxf(lmax, s);
  }
  red[tid] = lmax; __syncthreads();
  for (int st = 32; st > 0; st >>= 1) { if (tid < st) red[tid] = fmaxf(red[tid], red[tid + st]); __syncthreads(); }
  float gmax = red[0]; __syncthreads();
  float lsum = 0.f;
  for (int j = tid; j < NTOK; j += 64) { float e = __expf(sc[j] - gmax); sc[j] = e; lsum += e; }
  red[tid] = lsum; __syncthreads();
  for (int st = 32; st > 0; st >>= 1) { if (tid < st) red[tid] += red[tid + st]; __syncthreads(); }
  float inv = 1.f / red[0];
  __syncthreads();
  float acc = 0.f;
  for (int j = 0; j < NTOK; j++)
    acc += sc[j] * base[(long long)j * QKVW + 2 * D + h * DH + tid];
  out[(long long)(b * NTOK) * D + h * DH + tid] = f2bf(acc * inv);
}

// ---------------------------------------------------------------- time attention: per (b,h,nn), 4 queries x 5 keys
__global__ __launch_bounds__(64) void attn_time_k(const float* __restrict__ qkv,
                                                  __bf16* __restrict__ out) {
  int idx = blockIdx.x;
  int nn = idx % NP; idx /= NP;
  int h  = idx % H_HEADS;
  int b  = idx / H_HEADS;
  int tid = threadIdx.x;
  const float* base = qkv + (long long)b * NTOK * QKVW;
  __shared__ float qs[DH];
  __shared__ float sc[FR + 1];
  for (int ff = 0; ff < FR; ff++) {
    int qt = 1 + ff * NP + nn;
    __syncthreads();
    qs[tid] = base[(long long)qt * QKVW + h * DH + tid] * 0.125f;
    __syncthreads();
    if (tid < FR + 1) {
      int kt = (tid == 0) ? 0 : (1 + (tid - 1) * NP + nn);
      const float* kp = base + (long long)kt * QKVW + D + h * DH;
      float s = 0.f;
      for (int dd = 0; dd < DH; dd++) s += qs[dd] * kp[dd];
      sc[tid] = s;
    }
    __syncthreads();
    float m = sc[0];
    for (int j = 1; j <= FR; j++) m = fmaxf(m, sc[j]);
    float wgt[FR + 1];
    float sum = 0.f;
    for (int j = 0; j <= FR; j++) { wgt[j] = __expf(sc[j] - m); sum += wgt[j]; }
    float acc = 0.f;
    for (int j = 0; j <= FR; j++) {
      int kt = (j == 0) ? 0 : (1 + (j - 1) * NP + nn);
      acc += wgt[j] * base[(long long)kt * QKVW + 2 * D + h * DH + tid];
    }
    out[(long long)(b * NTOK + qt) * D + h * DH + tid] = f2bf(acc / sum);
  }
}

// ---------------------------------------------------------------- space attention: per (b,h,ff,nn), 197 keys
__global__ __launch_bounds__(64) void attn_space_k(const float* __restrict__ qkv,
                                                   __bf16* __restrict__ out) {
  int idx = blockIdx.x;
  int nn = idx % NP; idx /= NP;
  int ff = idx % FR; idx /= FR;
  int h  = idx % H_HEADS;
  int b  = idx / H_HEADS;
  int tid = threadIdx.x;
  int qt = 1 + ff * NP + nn;
  const float* base = qkv + (long long)b * NTOK * QKVW;
  __shared__ float qs[DH];
  __shared__ float sc[NP + 1];
  __shared__ float red[64];
  qs[tid] = base[(long long)qt * QKVW + h * DH + tid] * 0.125f;
  __syncthreads();
  float lmax = -3.4e38f;
  for (int j = tid; j < NP + 1; j += 64) {
    int kt = (j == 0) ? 0 : (1 + ff * NP + (j - 1));
    const float* kp = base + (long long)kt * QKVW + D + h * DH;
    float s = 0.f;
#pragma unroll 8
    for (int dd = 0; dd < DH; dd++) s += qs[dd] * kp[dd];
    sc[j] = s;
    lmax = fmaxf(lmax, s);
  }
  red[tid] = lmax; __syncthreads();
  for (int st = 32; st > 0; st >>= 1) { if (tid < st) red[tid] = fmaxf(red[tid], red[tid + st]); __syncthreads(); }
  float gmax = red[0]; __syncthreads();
  float lsum = 0.f;
  for (int j = tid; j < NP + 1; j += 64) { float e = __expf(sc[j] - gmax); sc[j] = e; lsum += e; }
  red[tid] = lsum; __syncthreads();
  for (int st = 32; st > 0; st >>= 1) { if (tid < st) red[tid] += red[tid + st]; __syncthreads(); }
  float inv = 1.f / red[0];
  __syncthreads();
  float acc = 0.f;
  for (int j = 0; j < NP + 1; j++) {
    int kt = (j == 0) ? 0 : (1 + ff * NP + (j - 1));
    acc += sc[j] * base[(long long)kt * QKVW + 2 * D + h * DH + tid];
  }
  out[(long long)(b * NTOK + qt) * D + h * DH + tid] = f2bf(acc * inv);
}

// ================================================================ host
extern "C" void kernel_launch(void* const* d_in, const int* in_sizes, int n_in,
                              void* d_out, int out_size, void* d_ws, size_t ws_size,
                              hipStream_t stream) {
  (void)in_sizes; (void)n_in; (void)out_size; (void)ws_size;
  const float* x       = (const float*)d_in[0];
  const float* patch_w = (const float*)d_in[1];
  const float* patch_b = (const float*)d_in[2];
  const float* cls_tok = (const float*)d_in[3];
  const float* pos_emb = (const float*)d_in[4];
  const float* tmp_emb = (const float*)d_in[5];
  const float* n1w = (const float*)d_in[6];
  const float* n1b = (const float*)d_in[7];
  const float* n2w = (const float*)d_in[8];
  const float* n2b = (const float*)d_in[9];
  const float* n3w = (const float*)d_in[10];
  const float* n3b = (const float*)d_in[11];
  const float* sqkv_w  = (const float*)d_in[12];
  const float* sqkv_b  = (const float*)d_in[13];
  const float* sproj_w = (const float*)d_in[14];
  const float* sproj_b = (const float*)d_in[15];
  const float* tqkv_w  = (const float*)d_in[16];
  const float* tqkv_b  = (const float*)d_in[17];
  const float* tproj_w = (const float*)d_in[18];
  const float* tproj_b = (const float*)d_in[19];
  const float* fc1_w = (const float*)d_in[20];
  const float* fc1_b = (const float*)d_in[21];
  const float* fc2_w = (const float*)d_in[22];
  const float* fc2_b = (const float*)d_in[23];
  const float* nw = (const float*)d_in[24];
  const float* nb = (const float*)d_in[25];
  const float* head_w = (const float*)d_in[26];
  const float* head_b = (const float*)d_in[27];

  char* ws = (char*)d_ws;
  size_t off = 0;
  auto alloc = [&](size_t bytes) -> void* {
    void* p = ws + off;
    off += (bytes + 255) & ~(size_t)255;
    return p;
  };

  __bf16* patchW = (__bf16*)alloc((size_t)D * D * 2);
  __bf16* sqkvW  = (__bf16*)alloc((size_t)NLAYER * QKVW * D * 2);
  __bf16* sprojW = (__bf16*)alloc((size_t)NLAYER * D * D * 2);
  __bf16* tqkvW  = (__bf16*)alloc((size_t)NLAYER * QKVW * D * 2);
  __bf16* tprojW = (__bf16*)alloc((size_t)NLAYER * D * D * 2);
  __bf16* fc1W   = (__bf16*)alloc((size_t)NLAYER * MLP_HID * D * 2);
  __bf16* fc2W   = (__bf16*)alloc((size_t)NLAYER * D * MLP_HID * 2);
  __bf16* headW  = (__bf16*)alloc((size_t)NCLS * D * 2);
  __bf16* Apatch = (__bf16*)alloc((size_t)BATCH * FR * NP * D * 2);
  float*  Pout   = (float*)alloc((size_t)BATCH * FR * NP * D * 4);
  float*  Hbuf   = (float*)alloc((size_t)MTOK * D * 4);
  float*  Tres   = (float*)alloc((size_t)MTOK * D * 4);
  float*  Sres   = (float*)alloc((size_t)MTOK * D * 4);
  __bf16* LnB    = (__bf16*)alloc((size_t)MTOK * D * 2);
  float*  Qkv    = (float*)alloc((size_t)MTOK * QKVW * 4);
  __bf16* AttnB  = (__bf16*)alloc((size_t)MTOK * D * 2);
  __bf16* MlpB   = (__bf16*)alloc((size_t)MTOK * MLP_HID * 2);

  auto cvt = [&](const float* src, __bf16* dst, size_t n) {
    cvt_k<<<dim3((unsigned)((n + 255) / 256)), 256, 0, stream>>>(src, dst, (long long)n);
  };
  cvt(patch_w, patchW, (size_t)D * D);
  cvt(sqkv_w,  sqkvW,  (size_t)NLAYER * QKVW * D);
  cvt(sproj_w, sprojW, (size_t)NLAYER * D * D);
  cvt(tqkv_w,  tqkvW,  (size_t)NLAYER * QKVW * D);
  cvt(tproj_w, tprojW, (size_t)NLAYER * D * D);
  cvt(fc1_w,   fc1W,   (size_t)NLAYER * MLP_HID * D);
  cvt(fc2_w,   fc2W,   (size_t)NLAYER * D * MLP_HID);
  cvt(head_w,  headW,  (size_t)NCLS * D);

  auto gemm = [&](const __bf16* A, long long lda, const __bf16* W, const float* bias,
                  const float* res, float* oF, __bf16* oB, int M, int N, int K, int gelu) {
    dim3 g((unsigned)((N + 63) / 64), (unsigned)((M + 63) / 64));
    gemm_bf16_k<<<g, 256, 0, stream>>>(A, lda, W, bias, res, oF, oB, M, N, K, gelu);
  };

  // Patch embedding
  {
    size_t n = (size_t)BATCH * FR * NP * D;
    im2col_k<<<dim3((unsigned)((n + 255) / 256)), 256, 0, stream>>>(x, Apatch);
    gemm(Apatch, D, patchW, patch_b, nullptr, Pout, nullptr, BATCH * FR * NP, D, D, 0);
    size_t m = (size_t)MTOK * D;
    assemble_h_k<<<dim3((unsigned)((m + 255) / 256)), 256, 0, stream>>>(Pout, cls_tok, pos_emb, tmp_emb, Hbuf);
  }

  for (int i = 0; i < NLAYER; i++) {
    // ---- time attention branch
    layernorm_k<<<MTOK, 256, 0, stream>>>(Hbuf, n3w + (size_t)i * D, n3b + (size_t)i * D, LnB);
    gemm(LnB, D, tqkvW + (size_t)i * QKVW * D, tqkv_b + (size_t)i * QKVW,
         nullptr, Qkv, nullptr, MTOK, QKVW, D, 0);
    attn_cls_k<<<BATCH * H_HEADS, 64, 0, stream>>>(Qkv, AttnB);
    attn_time_k<<<BATCH * H_HEADS * NP, 64, 0, stream>>>(Qkv, AttnB);
    gemm(AttnB, D, tprojW + (size_t)i * D * D, tproj_b + (size_t)i * D,
         Hbuf, Tres, nullptr, MTOK, D, D, 0);          // time_res = h + t_out

    // ---- space attention branch
    layernorm_k<<<MTOK, 256, 0, stream>>>(Tres, n1w + (size_t)i * D, n1b + (size_t)i * D, LnB);
    gemm(LnB, D, sqkvW + (size_t)i * QKVW * D, sqkv_b + (size_t)i * QKVW,
         nullptr, Qkv, nullptr, MTOK, QKVW, D, 0);
    attn_cls_k<<<BATCH * H_HEADS, 64, 0, stream>>>(Qkv, AttnB);
    attn_space_k<<<BATCH * H_HEADS * FR * NP, 64, 0, stream>>>(Qkv, AttnB);
    gemm(AttnB, D, sprojW + (size_t)i * D * D, sproj_b + (size_t)i * D,
         Hbuf, Sres, nullptr, MTOK, D, D, 0);          // space_res = h + s_out  (residual = layer input!)

    // ---- MLP
    layernorm_k<<<MTOK, 256, 0, stream>>>(Sres, n2w + (size_t)i * D, n2b + (size_t)i * D, LnB);
    gemm(LnB, D, fc1W + (size_t)i * MLP_HID * D, fc1_b + (size_t)i * MLP_HID,
         nullptr, nullptr, MlpB, MTOK, MLP_HID, D, 1); // GELU, bf16 out only
    gemm(MlpB, MLP_HID, fc2W + (size_t)i * D * MLP_HID, fc2_b + (size_t)i * D,
         Sres, Hbuf, nullptr, MTOK, D, MLP_HID, 0);    // h = space_res + z
  }

  // final LN + head (rows 0 and 785 = token 0 of each batch; lda skips a full sequence)
  layernorm_k<<<MTOK, 256, 0, stream>>>(Hbuf, nw, nb, LnB);
  gemm(LnB, (long long)NTOK * D, headW, head_b, nullptr, (float*)d_out, nullptr,
       BATCH, NCLS, D, 0);
}
